// LSTM_68066641707052
// MI455X (gfx1250) — matrix operand order repaired and checked by reading, not compiled, not run on hardware
//
#include <hip/hip_runtime.h>

#define VOCABSZ 4096
#define HIDSZ   1024
#define BATCHSZ 64
#define SEQLEN  256

typedef __bf16 bf16;
typedef __attribute__((ext_vector_type(16))) __bf16 v16bf;
typedef __attribute__((ext_vector_type(8)))  float  v8f;

__device__ __forceinline__ float sigmoidf_(float x) { return 1.f / (1.f + __expf(-x)); }
__device__ __forceinline__ float tanhf_(float x) {
    float e = __expf(2.f * x);
    return (e - 1.f) / (e + 1.f);
}

// ---------------------------------------------------------------------------
// Weight packing: f32 row-major [K x 4096] (N = g*1024 + h for gated weights)
// -> bf16 WMMA B-fragment layout [tileK][tileN(256)][lane(32)][e(16)]
//   lane l, elem e:  N = tileN*16 + (l & 15),  K = tileK*32 + (l>>4)*16 + e
// ---------------------------------------------------------------------------
__global__ void pack_gated(const float* __restrict__ src, bf16* __restrict__ dst,
                           int K, int srcRows, int koff) {
    long long tid = (long long)blockIdx.x * blockDim.x + threadIdx.x;
    long long total = (long long)K * 4096;
    if (tid >= total) return;
    int e    = (int)(tid & 15);
    int lane = (int)((tid >> 4) & 31);
    long long t = tid >> 9;
    int tileN = (int)(t % 256);
    int tileK = (int)(t / 256);
    int n = tileN * 16 + (lane & 15);
    int k = tileK * 32 + (lane >> 4) * 16 + e;
    int g = n >> 10;
    int h = n & 1023;
    float v = src[((long long)(g * srcRows + koff + k)) * HIDSZ + h];
    dst[tid] = (bf16)v;
}

__global__ void pack_plain(const float* __restrict__ src, bf16* __restrict__ dst) {
    long long tid = (long long)blockIdx.x * blockDim.x + threadIdx.x;
    if (tid >= (long long)HIDSZ * VOCABSZ) return;
    int e    = (int)(tid & 15);
    int lane = (int)((tid >> 4) & 31);
    long long t = tid >> 9;
    int tileN = (int)(t % 256);
    int tileK = (int)(t / 256);
    int n = tileN * 16 + (lane & 15);
    int k = tileK * 32 + (lane >> 4) * 16 + e;
    dst[tid] = (bf16)src[(long long)k * VOCABSZ + n];
}

// Zero the initial states (C1/C2 and the t=0 read-side packed H buffers).
__global__ void lstm_init(float* C1, float* C2, float* H1f, float* H2f,
                          bf16* H1P, bf16* H2P) {
    int i = blockIdx.x * blockDim.x + threadIdx.x;
    if (i < BATCHSZ * HIDSZ) {
        C1[i] = 0.f; C2[i] = 0.f; H1f[i] = 0.f; H2f[i] = 0.f;
        H1P[i] = (bf16)0.f; H1P[i + 65536] = (bf16)0.f;
        H2P[i] = (bf16)0.f; H2P[i + 65536] = (bf16)0.f;
    }
}

// ---------------------------------------------------------------------------
// A-fragment (16-bit 16x32) pack position for value at (row M=b, col K=h):
//   hb = kk>>4 ; e = kk&15 ; lane = (b&15) + 16*hb
// ---------------------------------------------------------------------------
__device__ __forceinline__ void pack_h_bf16(bf16* __restrict__ Hpack, int b, int h, float v) {
    int tileM = b >> 4, M = b & 15;
    int tileK = h >> 5, kk = h & 31;
    int hb = kk >> 4;
    int e  = kk & 15;
    int lane = M + 16 * hb;
    Hpack[(((long)tileK * 4 + tileM) * 32 + lane) * 16 + e] = (bf16)v;
}

// ---------------------------------------------------------------------------
// Layer 1: gates = H1_old @ W1rec + W1emb[tok] + b1 ; update C1, H1.
// grid = HID/16 workgroups (h-tile), 256 threads = 8 waves.
// wave w: gate g = w>>1, M-tiles {(w&1)*2, (w&1)*2+1}; 2 WMMA C tiles/wave.
// ---------------------------------------------------------------------------
__global__ __launch_bounds__(256)
void lstm_l1(const int* __restrict__ tokens, const float* __restrict__ W1,
             const float* __restrict__ b1, const bf16* __restrict__ W1P,
             const bf16* __restrict__ H1Pin, bf16* __restrict__ H1Pout,
             float* __restrict__ C1, float* __restrict__ H1f, int t) {
    __shared__ float gbuf[4][BATCHSZ][16];
    const int lane  = threadIdx.x & 31;
    const int wave  = threadIdx.x >> 5;
    const int g     = wave >> 1;
    const int m0    = (wave & 1) * 2;
    const int htile = blockIdx.x;
    const int ntile = g * 64 + htile;

    v8f c0 = {}; v8f c1 = {};
#pragma unroll 4
    for (int kt = 0; kt < 32; ++kt) {
        v16bf bm = *(const v16bf*)(W1P   + (((long)kt * 256 + ntile) * 32 + lane) * 16);
        v16bf a0 = *(const v16bf*)(H1Pin + (((long)kt * 4 + m0)     * 32 + lane) * 16);
        v16bf a1 = *(const v16bf*)(H1Pin + (((long)kt * 4 + m0 + 1) * 32 + lane) * 16);
        c0 = __builtin_amdgcn_wmma_f32_16x16x32_bf16(false, a0, false, bm, (short)0, c0, false, false);
        c1 = __builtin_amdgcn_wmma_f32_16x16x32_bf16(false, a1, false, bm, (short)0, c1, false, false);
    }
    const int nn = lane & 15;
    const int mh = (lane >> 4) * 8;
#pragma unroll
    for (int r = 0; r < 8; ++r) {
        gbuf[g][m0 * 16 + mh + r][nn]       = c0[r];
        gbuf[g][(m0 + 1) * 16 + mh + r][nn] = c1[r];
    }
    __syncthreads();
#pragma unroll
    for (int it = 0; it < 4; ++it) {
        int idx = it * 256 + threadIdx.x;   // 0..1023 -> (b, hh)
        int b  = idx >> 4;
        int hh = idx & 15;
        int h  = htile * 16 + hh;
        int tok = tokens[b * SEQLEN + t];
        float pre[4];
#pragma unroll
        for (int gg = 0; gg < 4; ++gg)
            pre[gg] = gbuf[gg][b][hh] + b1[gg * HIDSZ + h]
                    + W1[((long)(gg * (VOCABSZ + HIDSZ)) + tok) * HIDSZ + h];
        float fg = sigmoidf_(pre[0]);
        float ig = sigmoidf_(pre[1]);
        float og = sigmoidf_(pre[2]);
        float ct = tanhf_(pre[3]);
        float c  = fg * C1[b * HIDSZ + h] + ig * ct;
        C1[b * HIDSZ + h] = c;
        float Hv = og * tanhf_(c);
        H1f[b * HIDSZ + h] = Hv;
        pack_h_bf16(H1Pout, b, h, Hv);
    }
}

// ---------------------------------------------------------------------------
// Layer 2: gates = cat(H1_new, H2_old) @ W2 + b2 ; K = 2048 (64 k-tiles).
// ---------------------------------------------------------------------------
__global__ __launch_bounds__(256)
void lstm_l2(const bf16* __restrict__ W2P, const float* __restrict__ b2,
             const bf16* __restrict__ H1Pn, const bf16* __restrict__ H2Po,
             bf16* __restrict__ H2Pout, float* __restrict__ C2,
             float* __restrict__ H2f) {
    __shared__ float gbuf[4][BATCHSZ][16];
    const int lane  = threadIdx.x & 31;
    const int wave  = threadIdx.x >> 5;
    const int g     = wave >> 1;
    const int m0    = (wave & 1) * 2;
    const int htile = blockIdx.x;
    const int ntile = g * 64 + htile;

    v8f c0 = {}; v8f c1 = {};
#pragma unroll 4
    for (int kt = 0; kt < 64; ++kt) {
        const bf16* Abase = (kt < 32) ? (H1Pn + (long)kt * 2048)
                                      : (H2Po + (long)(kt - 32) * 2048);
        v16bf bm = *(const v16bf*)(W2P + (((long)kt * 256 + ntile) * 32 + lane) * 16);
        v16bf a0 = *(const v16bf*)(Abase + ((long)m0 * 32 + lane) * 16);
        v16bf a1 = *(const v16bf*)(Abase + ((long)(m0 + 1) * 32 + lane) * 16);
        c0 = __builtin_amdgcn_wmma_f32_16x16x32_bf16(false, a0, false, bm, (short)0, c0, false, false);
        c1 = __builtin_amdgcn_wmma_f32_16x16x32_bf16(false, a1, false, bm, (short)0, c1, false, false);
    }
    const int nn = lane & 15;
    const int mh = (lane >> 4) * 8;
#pragma unroll
    for (int r = 0; r < 8; ++r) {
        gbuf[g][m0 * 16 + mh + r][nn]       = c0[r];
        gbuf[g][(m0 + 1) * 16 + mh + r][nn] = c1[r];
    }
    __syncthreads();
#pragma unroll
    for (int it = 0; it < 4; ++it) {
        int idx = it * 256 + threadIdx.x;
        int b  = idx >> 4;
        int hh = idx & 15;
        int h  = htile * 16 + hh;
        float pre[4];
#pragma unroll
        for (int gg = 0; gg < 4; ++gg)
            pre[gg] = gbuf[gg][b][hh] + b2[gg * HIDSZ + h];
        float fg = sigmoidf_(pre[0]);
        float ig = sigmoidf_(pre[1]);
        float og = sigmoidf_(pre[2]);
        float ct = tanhf_(pre[3]);
        float c  = fg * C2[b * HIDSZ + h] + ig * ct;
        C2[b * HIDSZ + h] = c;
        float Hv = og * tanhf_(c);
        H2f[b * HIDSZ + h] = Hv;
        pack_h_bf16(H2Pout, b, h, Hv);
    }
}

// ---------------------------------------------------------------------------
// Output projection: Y[t] = H2_new @ Wout + bout -> d_out[t*B*V ...]
// grid = V/64 workgroups; wave w: n-tile = blockIdx*4 + (w>>1), M-tiles (w&1)*2..+1
// ---------------------------------------------------------------------------
__global__ __launch_bounds__(256)
void lstm_out(const bf16* __restrict__ WoutP, const bf16* __restrict__ H2Pn,
              const float* __restrict__ bout, float* __restrict__ out, int t) {
    const int lane  = threadIdx.x & 31;
    const int wave  = threadIdx.x >> 5;
    const int m0    = (wave & 1) * 2;
    const int ntile = blockIdx.x * 4 + (wave >> 1);

    v8f c0 = {}; v8f c1 = {};
#pragma unroll 4
    for (int kt = 0; kt < 32; ++kt) {
        v16bf bm = *(const v16bf*)(WoutP + (((long)kt * 256 + ntile) * 32 + lane) * 16);
        v16bf a0 = *(const v16bf*)(H2Pn  + (((long)kt * 4 + m0)     * 32 + lane) * 16);
        v16bf a1 = *(const v16bf*)(H2Pn  + (((long)kt * 4 + m0 + 1) * 32 + lane) * 16);
        c0 = __builtin_amdgcn_wmma_f32_16x16x32_bf16(false, a0, false, bm, (short)0, c0, false, false);
        c1 = __builtin_amdgcn_wmma_f32_16x16x32_bf16(false, a1, false, bm, (short)0, c1, false, false);
    }
    const int nn = lane & 15;
    const int mh = (lane >> 4) * 8;
    int n = ntile * 16 + nn;
    float bo = bout[n];
    long obase = ((long)t * BATCHSZ) * VOCABSZ + n;
#pragma unroll
    for (int r = 0; r < 8; ++r) {
        int ba = m0 * 16 + mh + r;
        int bb = (m0 + 1) * 16 + mh + r;
        out[obase + (long)ba * VOCABSZ] = c0[r] + bo;
        out[obase + (long)bb * VOCABSZ] = c1[r] + bo;
    }
}

// Final carry: out = (ys, H1, C1, H2, C2) concatenated flat.
__global__ void lstm_final(const float* __restrict__ H1f, const float* __restrict__ C1,
                           const float* __restrict__ H2f, const float* __restrict__ C2,
                           float* __restrict__ out) {
    int i = blockIdx.x * blockDim.x + threadIdx.x;
    long base = (long)SEQLEN * BATCHSZ * VOCABSZ;
    if (i < BATCHSZ * HIDSZ) {
        out[base + i]               = H1f[i];
        out[base + 65536 + i]       = C1[i];
        out[base + 2 * 65536 + i]   = H2f[i];
        out[base + 3 * 65536 + i]   = C2[i];
    }
}

extern "C" void kernel_launch(void* const* d_in, const int* in_sizes, int n_in,
                              void* d_out, int out_size, void* d_ws, size_t ws_size,
                              hipStream_t stream) {
    const int*   tokens = (const int*)d_in[0];
    const float* W1     = (const float*)d_in[1];
    const float* b1     = (const float*)d_in[2];
    const float* W2     = (const float*)d_in[3];
    const float* b2     = (const float*)d_in[4];
    const float* Wout   = (const float*)d_in[5];
    const float* bout   = (const float*)d_in[6];
    float* out = (float*)d_out;

    char* ws = (char*)d_ws;
    bf16* W1P   = (bf16*)(ws);                              //  8 MB : 1024x4096 bf16
    bf16* W2P   = (bf16*)(ws + (size_t)8  * 1024 * 1024);   // 16 MB : 2048x4096 bf16
    bf16* WoutP = (bf16*)(ws + (size_t)24 * 1024 * 1024);   //  8 MB : 1024x4096 bf16
    bf16* H1P   = (bf16*)(ws + (size_t)32 * 1024 * 1024);   // 2 x 64x1024 bf16 (double buffer)
    bf16* H2P   = H1P + 2 * 65536;
    float* C1   = (float*)(H2P + 2 * 65536);
    float* C2   = C1 + 65536;
    float* H1f  = C2 + 65536;
    float* H2f  = H1f + 65536;

    pack_gated<<<16384, 256, 0, stream>>>(W1, W1P, 1024, VOCABSZ + HIDSZ, VOCABSZ);
    pack_gated<<<32768, 256, 0, stream>>>(W2, W2P, 2048, 2 * HIDSZ, 0);
    pack_plain<<<16384, 256, 0, stream>>>(Wout, WoutP);
    lstm_init<<<256, 256, 0, stream>>>(C1, C2, H1f, H2f, H1P, H2P);

    for (int t = 0; t < SEQLEN; ++t) {
        int r = t & 1, w = r ^ 1;
        lstm_l1<<<HIDSZ / 16, 256, 0, stream>>>(tokens, W1, b1, W1P,
                                                H1P + r * 65536, H1P + w * 65536,
                                                C1, H1f, t);
        lstm_l2<<<HIDSZ / 16, 256, 0, stream>>>(W2P, b2,
                                                H1P + w * 65536, H2P + r * 65536,
                                                H2P + w * 65536, C2, H2f);
        lstm_out<<<VOCABSZ / 64, 256, 0, stream>>>(WoutP, H2P + w * 65536, bout, out, t);
    }
    lstm_final<<<256, 256, 0, stream>>>(H1f, C1, H2f, C2, out);
}
